// SuperPointLoRA_24764781429218
// MI455X (gfx1250) — matrix operand order repaired and checked by reading, not compiled, not run on hardware
//
#include <hip/hip_runtime.h>
#include <hip/hip_bf16.h>
#include <stdint.h>

typedef _Float16 half_t;
typedef __attribute__((ext_vector_type(16))) _Float16 v16h;
typedef __attribute__((ext_vector_type(8)))  float    v8f;

#define THRESH      0.0005f
#define CAP         16384
#define KTOP        2048
#define LORA_SCALE  0.0625f   // 1/RANK, RANK=16

// ---------------------------------------------------------------------------
// Weight packing: f32 OIHW -> f16 [tap][O][I]  (3x3)  /  [O_pad][I]  (1x1)
// ---------------------------------------------------------------------------
__global__ void weight_pack3_kern(const float* __restrict__ w, half_t* __restrict__ o,
                                  int CO, int CI, int total) {
    int e = blockIdx.x * 256 + threadIdx.x;
    if (e >= total) return;
    int ci  = e % CI;
    int t   = e / CI;
    int oc  = t % CO;
    int tap = t / CO;
    o[e] = (half_t)w[(size_t)(oc * CI + ci) * 9 + tap];
}

__global__ void weight_pack1_kern(const float* __restrict__ w, half_t* __restrict__ o,
                                  int CO, int CI, int total) {
    int e = blockIdx.x * 256 + threadIdx.x;
    if (e >= total) return;
    int ci = e % CI;
    int oc = e / CI;
    o[e] = (oc < CO) ? (half_t)w[(size_t)oc * CI + ci] : (half_t)0.0f;
}

// ---------------------------------------------------------------------------
// conv1a: 1 -> 64 channels, 3x3, relu, f32 image in, f16 out. K=9 only, so
// a direct LDS-tiled kernel beats WMMA setup cost here.
// ---------------------------------------------------------------------------
__global__ __launch_bounds__(256) void conv1a_kern(
    const float* __restrict__ img, const float* __restrict__ w,
    const float* __restrict__ bias, half_t* __restrict__ out, int H, int W) {
    __shared__ float t[18][18];
    __shared__ float ws[64 * 9];
    __shared__ float bs[64];
    int tx = threadIdx.x & 15, ty = threadIdx.x >> 4;
    int gx0 = blockIdx.x * 16, gy0 = blockIdx.y * 16;
    int b = blockIdx.z;
    const float* im = img + (size_t)b * H * W;
    for (int e = threadIdx.x; e < 18 * 18; e += 256) {
        int xx = e % 18, yy = e / 18;
        int gx = gx0 + xx - 1, gy = gy0 + yy - 1;
        t[yy][xx] = ((unsigned)gx < (unsigned)W && (unsigned)gy < (unsigned)H)
                        ? im[(size_t)gy * W + gx] : 0.0f;
    }
    for (int e = threadIdx.x; e < 576; e += 256) ws[e] = w[e];
    if (threadIdx.x < 64) bs[threadIdx.x] = bias[threadIdx.x];
    __syncthreads();
    float p[9];
#pragma unroll
    for (int ky = 0; ky < 3; ++ky)
#pragma unroll
        for (int kx = 0; kx < 3; ++kx) p[ky * 3 + kx] = t[ty + ky][tx + kx];
    int gx = gx0 + tx, gy = gy0 + ty;
    for (int oc = 0; oc < 64; ++oc) {
        float s = bs[oc];
#pragma unroll
        for (int k = 0; k < 9; ++k) s += ws[oc * 9 + k] * p[k];
        out[((size_t)b * 64 + oc) * H * W + (size_t)gy * W + gx] = (half_t)fmaxf(s, 0.0f);
    }
}

// ---------------------------------------------------------------------------
// 3x3 conv + bias + relu, implicit GEMM on v_wmma_f32_16x16x32_f16.
// Block = 256 thr = 8 waves; block tile = 32 px (one row) x 64 oc; each wave
// owns one 16x16 tile (px_t in {0,1}, oc_t in {0..3}).
// K loop: ci chunks of 32, 9 taps each -> one WMMA per (chunk,tap).
// ---------------------------------------------------------------------------
template <int CI, int CO>
__global__ __launch_bounds__(256) void conv3x3_wmma_kern(
    const half_t* __restrict__ in, const half_t* __restrict__ wpk,
    const float* __restrict__ bias, half_t* __restrict__ out, int H, int W) {
    __shared__ __attribute__((aligned(16))) half_t lw[9][64][32];  // [tap][oc][ci]
    __shared__ __attribute__((aligned(16))) half_t li[3][34][32];  // [ry][xx][ci]

    const int tid  = threadIdx.x;
    const int wv   = tid >> 5;
    const int lane = tid & 31;
    const int px_t = wv & 1;
    const int oc_t = wv >> 1;

    const int x0  = blockIdx.x * 32;
    const int y   = blockIdx.y;
    const int ocb = blockIdx.z % (CO / 64);
    const int b   = blockIdx.z / (CO / 64);
    const size_t HW = (size_t)H * W;

    const int m   = lane & 15;
    const int hic = lane >> 4;

    v8f acc = {};

    for (int cc = 0; cc < CI / 32; ++cc) {
        const half_t* inb = in + ((size_t)b * CI + cc * 32) * HW;
        for (int e = tid; e < 3 * 34 * 32; e += 256) {
            int ci = e & 31;
            int xx = (e >> 5) % 34;
            int ry = e / (34 * 32);
            int gy = y + ry - 1;
            int gx = x0 + xx - 1;
            half_t v = (half_t)0.0f;
            if ((unsigned)gy < (unsigned)H && (unsigned)gx < (unsigned)W)
                v = inb[(size_t)ci * HW + (size_t)gy * W + gx];
            li[ry][xx][ci] = v;
        }
        for (int e = tid; e < 9 * 64 * 32; e += 256) {
            int ci  = e & 31;
            int oc  = (e >> 5) & 63;
            int tap = e >> 11;
            lw[tap][oc][ci] = wpk[((size_t)tap * CO + ocb * 64 + oc) * CI + cc * 32 + ci];
        }
        __syncthreads();

#pragma unroll
        for (int tap = 0; tap < 9; ++tap) {
            const int ky = tap / 3, kx = tap % 3;
            union { v16h h; uint32_t u[8]; } fa, fb;
#pragma unroll
            for (int j = 0; j < 8; ++j) {
                int k0 = hic * 16 + 2 * j;                                 // A k order used here: contiguous, k = 16*hic + s
                fa.u[j] = *(const uint32_t*)&lw[tap][oc_t * 16 + m][k0];
            }
#pragma unroll
            for (int j = 0; j < 8; ++j) {
                int k0 = hic * 16 + 2 * j;                                 // B k order used here: contiguous, same as A
                fb.u[j] = *(const uint32_t*)&li[ky][px_t * 16 + m + kx][k0];
            }
            acc = __builtin_amdgcn_wmma_f32_16x16x32_f16(
                false, fa.h, false, fb.h, (short)0, acc, false, false);
        }
        __syncthreads();
    }

#pragma unroll
    for (int r = 0; r < 8; ++r) {
        int oc = ocb * 64 + oc_t * 16 + r + hic * 8;   // C/D layout: VGPR r -> M=r / r+8
        int px = x0 + px_t * 16 + m;
        float v = acc[r] + bias[oc];
        v = fmaxf(v, 0.0f);
        out[((size_t)b * CO + oc) * HW + (size_t)y * W + px] = (half_t)v;
    }
}

// ---------------------------------------------------------------------------
// 1x1 conv (GEMM) on WMMA. CO is the padded/stride channel count; stores are
// guarded by CO_VALID (score head: 65 of 128).
// ---------------------------------------------------------------------------
template <int CI, int CO, int CO_VALID, bool RELU, bool OUTF32>
__global__ __launch_bounds__(256) void conv1x1_wmma_kern(
    const half_t* __restrict__ in, const half_t* __restrict__ w,
    const float* __restrict__ bias, void* __restrict__ outv, int HW) {
    __shared__ __attribute__((aligned(16))) half_t lw[64][32];  // [oc][ci]
    __shared__ __attribute__((aligned(16))) half_t li[32][32];  // [n][ci]

    const int tid  = threadIdx.x;
    const int wv   = tid >> 5;
    const int lane = tid & 31;
    const int px_t = wv & 1;
    const int oc_t = wv >> 1;

    const int p0  = blockIdx.x * 32;
    const int ocb = blockIdx.y;
    const int b   = blockIdx.z;

    const int m   = lane & 15;
    const int hic = lane >> 4;

    v8f acc = {};

    for (int cc = 0; cc < CI / 32; ++cc) {
        for (int e = tid; e < 32 * 32; e += 256) {
            int n = e & 31, ci = e >> 5;
            li[n][ci] = in[((size_t)b * CI + cc * 32 + ci) * HW + p0 + n];
        }
        for (int e = tid; e < 64 * 32; e += 256) {
            int ci = e & 31, oc = e >> 5;
            lw[oc][ci] = w[(size_t)(ocb * 64 + oc) * CI + cc * 32 + ci];
        }
        __syncthreads();

        union { v16h h; uint32_t u[8]; } fa, fb;
#pragma unroll
        for (int j = 0; j < 8; ++j) {
            int k0 = hic * 16 + 2 * j;
            fa.u[j] = *(const uint32_t*)&lw[oc_t * 16 + m][k0];
        }
#pragma unroll
        for (int j = 0; j < 8; ++j) {
            int k0 = hic * 16 + 2 * j;
            fb.u[j] = *(const uint32_t*)&li[px_t * 16 + m][k0];
        }
        acc = __builtin_amdgcn_wmma_f32_16x16x32_f16(
            false, fa.h, false, fb.h, (short)0, acc, false, false);
        __syncthreads();
    }

#pragma unroll
    for (int r = 0; r < 8; ++r) {
        int oc = ocb * 64 + oc_t * 16 + r + hic * 8;
        int px = p0 + px_t * 16 + m;
        if (oc < CO_VALID) {
            float v = acc[r] + bias[oc];
            if (RELU) v = fmaxf(v, 0.0f);
            if (OUTF32)
                ((float*)outv)[((size_t)b * CO + oc) * HW + px] = v;
            else
                ((half_t*)outv)[((size_t)b * CO + oc) * HW + px] = (half_t)v;
        }
    }
}

// ---------------------------------------------------------------------------
// 2x2 max pool (f16)
// ---------------------------------------------------------------------------
__global__ void maxpool2_kern(const half_t* __restrict__ in, half_t* __restrict__ out,
                              int C, int Ho, int Wo, int total) {
    int i = blockIdx.x * 256 + threadIdx.x;
    if (i >= total) return;
    int xo = i % Wo;
    int t  = i / Wo;
    int yo = t % Ho; t /= Ho;
    int c  = t % C;
    int b  = t / C;
    int Hi = Ho * 2, Wi = Wo * 2;
    const half_t* p = in + (size_t)(b * C + c) * Hi * Wi;
    float a = (float)p[(size_t)(2 * yo) * Wi + 2 * xo];
    a = fmaxf(a, (float)p[(size_t)(2 * yo) * Wi + 2 * xo + 1]);
    a = fmaxf(a, (float)p[(size_t)(2 * yo + 1) * Wi + 2 * xo]);
    a = fmaxf(a, (float)p[(size_t)(2 * yo + 1) * Wi + 2 * xo + 1]);
    out[i] = (half_t)a;
}

// ---------------------------------------------------------------------------
// softmax over 65 ch (drop dustbin) + 8x8 pixel-shuffle into 512x512 map
// ---------------------------------------------------------------------------
__global__ void softmax_shuffle_kern(const float* __restrict__ sc, float* __restrict__ smap) {
    int i = blockIdx.x * 256 + threadIdx.x;
    if (i >= 2 * 4096) return;
    int b = i >> 12;
    int pix = i & 4095;
    int y = pix >> 6, x = pix & 63;
    const float* p = sc + (size_t)b * 128 * 4096 + pix;
    float mx = -1e30f;
    for (int c = 0; c < 65; ++c) mx = fmaxf(mx, p[(size_t)c * 4096]);
    float sum = 0.f;
    for (int c = 0; c < 65; ++c) sum += expf(p[(size_t)c * 4096] - mx);
    float inv = 1.0f / sum;
    float* o = smap + (size_t)b * 512 * 512;
    for (int c = 0; c < 64; ++c) {
        float v = expf(p[(size_t)c * 4096] - mx) * inv;
        int dy = c >> 3, dx = c & 7;
        o[(size_t)(y * 8 + dy) * 512 + (x * 8 + dx)] = v;
    }
}

// ---------------------------------------------------------------------------
// separable 9x9 max filter (NMS pooling), -inf padding
// ---------------------------------------------------------------------------
__global__ void maxfilt_h_kern(const float* __restrict__ in, float* __restrict__ out, int n) {
    int i = blockIdx.x * 256 + threadIdx.x;
    if (i >= n) return;
    int x = i % 512;
    int base = i - x;
    float m = -3.0e38f;
    for (int d = -4; d <= 4; ++d) {
        int xx = x + d;
        if ((unsigned)xx < 512u) m = fmaxf(m, in[base + xx]);
    }
    out[i] = m;
}

__global__ void maxfilt_v_kern(const float* __restrict__ in, float* __restrict__ out, int n) {
    int i = blockIdx.x * 256 + threadIdx.x;
    if (i >= n) return;
    int x = i % 512;
    int y = (i / 512) % 512;
    const float* p = in + (size_t)(i / (512 * 512)) * 512 * 512;
    float m = -3.0e38f;
    for (int d = -4; d <= 4; ++d) {
        int yy = y + d;
        if ((unsigned)yy < 512u) m = fmaxf(m, p[(size_t)yy * 512 + x]);
    }
    out[i] = m;
}

__global__ void eq_mask_kern(const float* s, const float* p, float* m, int n) {
    int i = blockIdx.x * 256 + threadIdx.x;
    if (i < n) m[i] = (s[i] == p[i]) ? 1.0f : 0.0f;
}
__global__ void suppress_kern(const float* s, const float* pm, float* ss, int n) {
    int i = blockIdx.x * 256 + threadIdx.x;
    if (i < n) ss[i] = (pm[i] > 0.0f) ? 0.0f : s[i];
}
__global__ void mask_update_kern(float* m, const float* ss, const float* pss,
                                 const float* pm, int n) {
    int i = blockIdx.x * 256 + threadIdx.x;
    if (i < n) {
        bool nm = (ss[i] == pss[i]) && !(pm[i] > 0.0f);
        if (nm) m[i] = 1.0f;
    }
}
__global__ void finalize_kern(const float* s, const float* m, float* f, int n) {
    int i = blockIdx.x * 256 + threadIdx.x;
    if (i >= n) return;
    int x = i % 512;
    int y = (i / 512) % 512;
    float v = (m[i] > 0.0f) ? s[i] : 0.0f;
    if (x < 4 || x >= 508 || y < 4 || y >= 508) v = -1.0f;
    f[i] = v;
}

// ---------------------------------------------------------------------------
// top-K: compact survivors > THRESH into packed (floatbits<<32)|idx keys,
// single-block bitonic sort (descending) per batch.
// ---------------------------------------------------------------------------
__global__ void topk_init_kern(unsigned long long* keys, int* cnt) {
    int i = blockIdx.x * 256 + threadIdx.x;
    if (i < 2 * CAP) keys[i] = 0ull;
    if (i < 2) cnt[i] = 0;
}
__global__ void compact_kern(const float* __restrict__ F, unsigned long long* keys,
                             int* cnt, int n) {
    int i = blockIdx.x * 256 + threadIdx.x;
    if (i >= n) return;
    float v = F[i];
    if (v > THRESH) {
        int b = i / (512 * 512);
        int idx = i - b * 512 * 512;
        int pos = atomicAdd(&cnt[b], 1);
        if (pos < CAP)
            keys[(size_t)b * CAP + pos] =
                ((unsigned long long)__float_as_uint(v) << 32) | (unsigned)idx;
    }
}
__global__ __launch_bounds__(1024) void bitonic_kern(unsigned long long* keys_all) {
    unsigned long long* keys = keys_all + (size_t)blockIdx.x * CAP;
    for (int k = 2; k <= CAP; k <<= 1) {
        for (int j = k >> 1; j > 0; j >>= 1) {
            for (int i = threadIdx.x; i < CAP; i += 1024) {
                int ixj = i ^ j;
                if (ixj > i) {
                    unsigned long long a = keys[i], c = keys[ixj];
                    bool desc = ((i & k) == 0);
                    bool sw = desc ? (a < c) : (a > c);
                    if (sw) { keys[i] = c; keys[ixj] = a; }
                }
            }
            __syncthreads();
        }
    }
}
__global__ void emit_kern(const unsigned long long* __restrict__ keys, float* __restrict__ out) {
    int i = blockIdx.x * 256 + threadIdx.x;
    if (i >= 2 * KTOP) return;
    int b = i / KTOP, j = i % KTOP;
    unsigned long long key = keys[(size_t)b * CAP + j];
    float v = __uint_as_float((unsigned)(key >> 32));
    unsigned idx = (unsigned)key;
    float kx = (float)(idx % 512u), ky = (float)(idx / 512u);
    bool valid = v > THRESH;
    out[(size_t)i * 2 + 0] = valid ? kx : 0.0f;
    out[(size_t)i * 2 + 1] = valid ? ky : 0.0f;
    out[2 * KTOP * 2 + i]  = valid ? v : 0.0f;
}

// ---------------------------------------------------------------------------
// LoRA: y = x + B(Ax) * 1/16, per pixel (C=256, rank 16). Block = one pixel.
// ---------------------------------------------------------------------------
template <typename T>
__global__ __launch_bounds__(256) void lora_add_kern(
    const T* __restrict__ in, const float* __restrict__ A,
    const float* __restrict__ Bm, T* __restrict__ out) {
    __shared__ float xs[256];
    __shared__ float tr[16];
    int c = threadIdx.x;
    int pix = blockIdx.x;
    int b = blockIdx.y;
    size_t off = ((size_t)b * 256 + c) * 4096 + pix;
    float x = (float)in[off];
    xs[c] = x;
    __syncthreads();
    if (c < 16) {
        float s = 0.0f;
        for (int i = 0; i < 256; ++i) s += A[c * 256 + i] * xs[i];
        tr[c] = s;
    }
    __syncthreads();
    float l = 0.0f;
#pragma unroll
    for (int r = 0; r < 16; ++r) l += Bm[c * 16 + r] * tr[r];
    out[off] = (T)(x + l * LORA_SCALE);
}

// ---------------------------------------------------------------------------
// per-pixel channel L2 norm of descriptor map (f32)
// ---------------------------------------------------------------------------
__global__ __launch_bounds__(256) void l2norm_map_kern(const float* __restrict__ in,
                                                       float* __restrict__ out) {
    __shared__ float red[256];
    int c = threadIdx.x;
    int pix = blockIdx.x;
    int b = blockIdx.y;
    float v = in[((size_t)b * 256 + c) * 4096 + pix];
    red[c] = v * v;
    __syncthreads();
    for (int s = 128; s > 0; s >>= 1) {
        if (c < s) red[c] += red[c + s];
        __syncthreads();
    }
    float nrm = fmaxf(sqrtf(red[0]), 1e-12f);
    out[((size_t)b * 256 + c) * 4096 + pix] = v / nrm;
}

// ---------------------------------------------------------------------------
// bilinear sample + L2 norm + validity mask; writes dsc (b,K,256) region
// ---------------------------------------------------------------------------
__global__ __launch_bounds__(256) void sample_kern(
    const unsigned long long* __restrict__ keys, const float* __restrict__ descn,
    float* __restrict__ out) {
    __shared__ float red[256];
    int c = threadIdx.x;
    int kp = blockIdx.x;
    int b = blockIdx.y;
    unsigned long long key = keys[(size_t)b * CAP + kp];
    float v = __uint_as_float((unsigned)(key >> 32));
    unsigned idx = (unsigned)key;
    float kx = (float)(idx % 512u), ky = (float)(idx / 512u);
    bool valid = v > THRESH;
    // g = ((k - 3.5)/507.5)*2 - 1 ;  x = (g+1)*0.5*63
    float gx = ((kx - 3.5f) / 507.5f) * 2.0f - 1.0f;
    float gy = ((ky - 3.5f) / 507.5f) * 2.0f - 1.0f;
    float x = (gx + 1.0f) * 0.5f * 63.0f;
    float y = (gy + 1.0f) * 0.5f * 63.0f;
    x = fminf(fmaxf(x, 0.0f), 63.0f);
    y = fminf(fmaxf(y, 0.0f), 63.0f);
    float x0f = floorf(x), y0f = floorf(y);
    float wx = x - x0f, wy = y - y0f;
    int x0 = (int)x0f; if (x0 < 0) x0 = 0; if (x0 > 63) x0 = 63;
    int y0 = (int)y0f; if (y0 < 0) y0 = 0; if (y0 > 63) y0 = 63;
    int x1 = x0 + 1; if (x1 > 63) x1 = 63;
    int y1 = y0 + 1; if (y1 > 63) y1 = 63;
    const float* p = descn + ((size_t)b * 256 + c) * 4096;
    float d = p[y0 * 64 + x0] * (1 - wx) * (1 - wy) + p[y0 * 64 + x1] * wx * (1 - wy) +
              p[y1 * 64 + x0] * (1 - wx) * wy       + p[y1 * 64 + x1] * wx * wy;
    red[c] = d * d;
    __syncthreads();
    for (int s = 128; s > 0; s >>= 1) {
        if (c < s) red[c] += red[c + s];
        __syncthreads();
    }
    float nrm = fmaxf(sqrtf(red[0]), 1e-12f);
    float o = valid ? (d / nrm) : 0.0f;
    out[2 * KTOP * 2 + 2 * KTOP + ((size_t)(b * KTOP + kp)) * 256 + c] = o;
}

// ---------------------------------------------------------------------------
// host orchestration
// ---------------------------------------------------------------------------
extern "C" void kernel_launch(void* const* d_in, const int* in_sizes, int n_in,
                              void* d_out, int out_size, void* d_ws, size_t ws_size,
                              hipStream_t stream) {
    (void)in_sizes; (void)n_in; (void)out_size; (void)ws_size;
    const float* image = (const float*)d_in[0];
    const float* w1a = (const float*)d_in[1];  const float* b1a = (const float*)d_in[2];
    const float* w1b = (const float*)d_in[3];  const float* b1b = (const float*)d_in[4];
    const float* w2a = (const float*)d_in[5];  const float* b2a = (const float*)d_in[6];
    const float* w2b = (const float*)d_in[7];  const float* b2b = (const float*)d_in[8];
    const float* w3a = (const float*)d_in[9];  const float* b3a = (const float*)d_in[10];
    const float* w3b = (const float*)d_in[11]; const float* b3b = (const float*)d_in[12];
    const float* w4a = (const float*)d_in[13]; const float* b4a = (const float*)d_in[14];
    const float* w4b = (const float*)d_in[15]; const float* b4b = (const float*)d_in[16];
    const float* wPa = (const float*)d_in[17]; const float* bPa = (const float*)d_in[18];
    const float* wPb = (const float*)d_in[19]; const float* bPb = (const float*)d_in[20];
    const float* wDa = (const float*)d_in[21]; const float* bDa = (const float*)d_in[22];
    const float* wDb = (const float*)d_in[23]; const float* bDb = (const float*)d_in[24];
    const float* laA = (const float*)d_in[25]; const float* laB = (const float*)d_in[26];
    const float* lbA = (const float*)d_in[27]; const float* lbB = (const float*)d_in[28];

    char* wsb = (char*)d_ws;
    size_t off = 0;
    auto alloc = [&](size_t bytes) -> char* {
        char* p = wsb + off;
        off += (bytes + 255) & ~(size_t)255;
        return p;
    };

    half_t* P0 = (half_t*)alloc((size_t)2 * 64 * 512 * 512 * 2);
    half_t* P1 = (half_t*)alloc((size_t)2 * 64 * 512 * 512 * 2);
    float*  SC65  = (float*)alloc((size_t)2 * 128 * 4096 * 4);
    float*  SM_s  = (float*)alloc((size_t)2 * 512 * 512 * 4);
    float*  SM_t  = (float*)alloc((size_t)2 * 512 * 512 * 4);
    float*  SM_p1 = (float*)alloc((size_t)2 * 512 * 512 * 4);
    float*  SM_p2 = (float*)alloc((size_t)2 * 512 * 512 * 4);
    float*  SM_m  = (float*)alloc((size_t)2 * 512 * 512 * 4);
    float*  SM_f  = (float*)alloc((size_t)2 * 512 * 512 * 4);
    float*  DESCF = (float*)alloc((size_t)2 * 256 * 4096 * 4);
    float*  DESCN = (float*)alloc((size_t)2 * 256 * 4096 * 4);
    unsigned long long* KEYS = (unsigned long long*)alloc((size_t)2 * CAP * 8);
    int* CNT = (int*)alloc(256);
    half_t* Pw1b = (half_t*)alloc((size_t)9 * 64 * 64 * 2);
    half_t* Pw2a = (half_t*)alloc((size_t)9 * 64 * 64 * 2);
    half_t* Pw2b = (half_t*)alloc((size_t)9 * 64 * 64 * 2);
    half_t* Pw3a = (half_t*)alloc((size_t)9 * 128 * 64 * 2);
    half_t* Pw3b = (half_t*)alloc((size_t)9 * 128 * 128 * 2);
    half_t* Pw4a = (half_t*)alloc((size_t)9 * 128 * 128 * 2);
    half_t* Pw4b = (half_t*)alloc((size_t)9 * 128 * 128 * 2);
    half_t* PwPa = (half_t*)alloc((size_t)9 * 256 * 128 * 2);
    half_t* PwDa = (half_t*)alloc((size_t)9 * 256 * 128 * 2);
    half_t* PwPb = (half_t*)alloc((size_t)128 * 256 * 2);
    half_t* PwDb = (half_t*)alloc((size_t)256 * 256 * 2);

    auto pk3 = [&](const float* w, half_t* o, int CO, int CI) {
        int tot = 9 * CO * CI;
        weight_pack3_kern<<<(tot + 255) / 256, 256, 0, stream>>>(w, o, CO, CI, tot);
    };
    pk3(w1b, Pw1b, 64, 64);   pk3(w2a, Pw2a, 64, 64);   pk3(w2b, Pw2b, 64, 64);
    pk3(w3a, Pw3a, 128, 64);  pk3(w3b, Pw3b, 128, 128);
    pk3(w4a, Pw4a, 128, 128); pk3(w4b, Pw4b, 128, 128);
    pk3(wPa, PwPa, 256, 128); pk3(wDa, PwDa, 256, 128);
    {
        int tot = 128 * 256;
        weight_pack1_kern<<<(tot + 255) / 256, 256, 0, stream>>>(wPb, PwPb, 65, 256, tot);
        tot = 256 * 256;
        weight_pack1_kern<<<(tot + 255) / 256, 256, 0, stream>>>(wDb, PwDb, 256, 256, tot);
    }

    // ---- backbone ----
    conv1a_kern<<<dim3(32, 32, 2), 256, 0, stream>>>(image, w1a, b1a, P0, 512, 512);
    conv3x3_wmma_kern<64, 64><<<dim3(16, 512, 2), 256, 0, stream>>>(P0, Pw1b, b1b, P1, 512, 512);
    { int t = 2 * 64 * 256 * 256; maxpool2_kern<<<(t + 255) / 256, 256, 0, stream>>>(P1, P0, 64, 256, 256, t); }
    conv3x3_wmma_kern<64, 64><<<dim3(8, 256, 2), 256, 0, stream>>>(P0, Pw2a, b2a, P1, 256, 256);
    conv3x3_wmma_kern<64, 64><<<dim3(8, 256, 2), 256, 0, stream>>>(P1, Pw2b, b2b, P0, 256, 256);
    { int t = 2 * 64 * 128 * 128; maxpool2_kern<<<(t + 255) / 256, 256, 0, stream>>>(P0, P1, 64, 128, 128, t); }
    conv3x3_wmma_kern<64, 128><<<dim3(4, 128, 4), 256, 0, stream>>>(P1, Pw3a, b3a, P0, 128, 128);
    conv3x3_wmma_kern<128, 128><<<dim3(4, 128, 4), 256, 0, stream>>>(P0, Pw3b, b3b, P1, 128, 128);
    { int t = 2 * 128 * 64 * 64; maxpool2_kern<<<(t + 255) / 256, 256, 0, stream>>>(P1, P0, 128, 64, 64, t); }
    conv3x3_wmma_kern<128, 128><<<dim3(2, 64, 4), 256, 0, stream>>>(P0, Pw4a, b4a, P1, 64, 64);
    conv3x3_wmma_kern<128, 128><<<dim3(2, 64, 4), 256, 0, stream>>>(P1, Pw4b, b4b, P0, 64, 64);
    // P0 = x (x4b), f16, 2x128x64x64

    // ---- score head ----
    conv3x3_wmma_kern<128, 256><<<dim3(2, 64, 8), 256, 0, stream>>>(P0, PwPa, bPa, P1, 64, 64);
    conv1x1_wmma_kern<256, 128, 65, false, true>
        <<<dim3(128, 2, 2), 256, 0, stream>>>(P1, PwPb, bPb, SC65, 4096);

    // ---- descriptor head ----
    conv3x3_wmma_kern<128, 256><<<dim3(2, 64, 8), 256, 0, stream>>>(P0, PwDa, bDa, P1, 64, 64);
    lora_add_kern<half_t><<<dim3(4096, 2), 256, 0, stream>>>(P1, laA, laB, P0);
    conv1x1_wmma_kern<256, 256, 256, false, true>
        <<<dim3(128, 4, 2), 256, 0, stream>>>(P0, PwDb, bDb, DESCF, 4096);
    lora_add_kern<float><<<dim3(4096, 2), 256, 0, stream>>>(DESCF, lbA, lbB, DESCF);
    l2norm_map_kern<<<dim3(4096, 2), 256, 0, stream>>>(DESCF, DESCN);

    // ---- scores: softmax + shuffle + NMS + border + top-K ----
    softmax_shuffle_kern<<<32, 256, 0, stream>>>(SC65, SM_s);
    const int NT = 2 * 512 * 512;
    const int GB = NT / 256;
    auto pool9 = [&](const float* src, float* dst) {
        maxfilt_h_kern<<<GB, 256, 0, stream>>>(src, SM_t, NT);
        maxfilt_v_kern<<<GB, 256, 0, stream>>>(SM_t, dst, NT);
    };
    pool9(SM_s, SM_p1);
    eq_mask_kern<<<GB, 256, 0, stream>>>(SM_s, SM_p1, SM_m, NT);
    for (int it = 0; it < 2; ++it) {
        pool9(SM_m, SM_p1);
        suppress_kern<<<GB, 256, 0, stream>>>(SM_s, SM_p1, SM_f, NT);
        pool9(SM_f, SM_p2);
        mask_update_kern<<<GB, 256, 0, stream>>>(SM_m, SM_f, SM_p2, SM_p1, NT);
    }
    finalize_kern<<<GB, 256, 0, stream>>>(SM_s, SM_m, SM_f, NT);

    topk_init_kern<<<(2 * CAP + 255) / 256, 256, 0, stream>>>(KEYS, CNT);
    compact_kern<<<GB, 256, 0, stream>>>(SM_f, KEYS, CNT, NT);
    bitonic_kern<<<2, 1024, 0, stream>>>(KEYS);
    emit_kern<<<(2 * KTOP + 255) / 256, 256, 0, stream>>>(KEYS, (float*)d_out);
    sample_kern<<<dim3(KTOP, 2), 256, 0, stream>>>(KEYS, DESCN, (float*)d_out);
}
